// feature_Attention_CostAgg_Module_65335042506914
// MI455X (gfx1250) — matrix operand order repaired and checked by reading, not compiled, not run on hardware
//
#include <hip/hip_runtime.h>

// ---------------- problem constants ----------------
#define Bb 4
#define Cc 32
#define Dd 64
#define Hh 128
#define Ww 256
#define HWp (Hh * Ww)
#define BN_EPS 1e-5f

typedef __bf16 bf16_t;
typedef __attribute__((ext_vector_type(16))) __bf16 v16bf;
typedef __attribute__((ext_vector_type(8)))  __bf16 v8bf;
typedef __attribute__((ext_vector_type(8)))  float  v8f;

// ---------------- WMMA fragment helpers (CDNA5 16-bit layouts, ISA 7.12.2) ----
// A (16x32, MxK): lane m=lane&15, half=lane>>4; elems 0..7 = K(8h..8h+7),
//                 elems 8..15 = K(16+8h..16+8h+7).
__device__ __forceinline__ v16bf load_frag_a(const bf16_t* base, int stride, int k0) {
  const int lane = threadIdx.x & 31;
  const bf16_t* p = base + (size_t)(lane & 15) * stride + k0 + ((lane >> 4) << 3);
  union { v16bf v; v8bf h[2]; } f;
  f.h[0] = *(const v8bf*)p;
  f.h[1] = *(const v8bf*)(p + 16);
  return f.v;
}
// B (32x16, KxN) stored as [N][K]-contiguous: lane n=lane&15, half=lane>>4;
// elems 0..7 = K(8h..8h+7), elems 8..15 = K(16+8h..16+8h+7).
__device__ __forceinline__ v16bf load_frag_b(const bf16_t* base, int stride, int k0) {
  const int lane = threadIdx.x & 31;
  const bf16_t* p = base + (size_t)(lane & 15) * stride + k0 + ((lane >> 4) << 3);
  union { v16bf v; v8bf h[2]; } f;
  f.h[0] = *(const v8bf*)p;
  f.h[1] = *(const v8bf*)(p + 16);
  return f.v;
}
__device__ __forceinline__ v8f wmma_bf16(v16bf a, v16bf b, v8f c) {
  return __builtin_amdgcn_wmma_f32_16x16x32_bf16(false, a, false, b, (short)0, c,
                                                 false, false);
}

// ---------------- workspace layout (bytes) ----------------
// eH/eW now bf16 (logits written bf16, softmax'd in place): halves the
// dominant HBM traffic (~200 MB saved vs f32 logits).
static constexpr size_t SZ_T   = (size_t)Bb * Dd * HWp * 4;            // 33.5 MB
static constexpr size_t OFF_T0 = 0;
static constexpr size_t OFF_T1 = OFF_T0 + SZ_T;
static constexpr size_t OFF_OH = OFF_T1 + SZ_T;
static constexpr size_t OFF_EH = OFF_OH + SZ_T;                        // B*W*H*H bf16
static constexpr size_t OFF_EW = OFF_EH + (size_t)Bb * Ww * Hh * Hh * 2;
static constexpr size_t OFF_Q  = OFF_EW + (size_t)Bb * Hh * Ww * Ww * 2;
static constexpr size_t OFF_K  = OFF_Q + (size_t)Bb * Cc * HWp * 2;
static constexpr size_t OFF_V  = OFF_K + (size_t)Bb * Cc * HWp * 2;
static constexpr size_t OFF_WA = OFF_V + (size_t)Bb * Dd * HWp * 2;
static constexpr size_t OFF_WB = OFF_WA + 36864 * 2;
static constexpr size_t OFF_WQ = OFF_WB + 36864 * 2;
static constexpr size_t OFF_WK = OFF_WQ + 1024 * 2;
static constexpr size_t OFF_WV = OFF_WK + 1024 * 2;
static constexpr size_t OFF_ST = OFF_WV + 4096 * 2;                    // 128 f32
static constexpr size_t OFF_PT = OFF_ST + 512;                         // 2048 f32

// ---------------- weight conversion to bf16 ----------------
// wa/wb layout: [tap(9)][oc(64)][ic(64)] ; wq/wk: [oc(32)][ic(32)] ; wv: [oc][ic]
__global__ void k_wconv(const float* __restrict__ Wa, const float* __restrict__ Wb,
                        const float* __restrict__ Wq, const float* __restrict__ Wk,
                        const float* __restrict__ Wv, bf16_t* __restrict__ wa,
                        bf16_t* __restrict__ wb, bf16_t* __restrict__ wq,
                        bf16_t* __restrict__ wk, bf16_t* __restrict__ wv) {
  int i = blockIdx.x * 256 + threadIdx.x;
  if (i < 36864) {
    int tap = i >> 12, rem = i & 4095, o = rem >> 6, ic = rem & 63;
    wa[i] = (bf16_t)Wa[((o * 64 + ic) * 3 + tap / 3) * 3 + tap % 3];
  } else if (i < 73728) {
    int j = i - 36864;
    int tap = j >> 12, rem = j & 4095, o = rem >> 6, ic = rem & 63;
    wb[j] = (bf16_t)Wb[((o * 64 + ic) * 3 + tap / 3) * 3 + tap % 3];
  } else if (i < 74752) {
    wq[i - 73728] = (bf16_t)Wq[i - 73728];
  } else if (i < 75776) {
    wk[i - 74752] = (bf16_t)Wk[i - 74752];
  } else if (i < 79872) {
    wv[i - 75776] = (bf16_t)Wv[i - 75776];
  }
}

// ---------------- 3x3 conv, implicit GEMM via WMMA ----------------
// grid (B*H, W/32), 128 threads = 4 waves; wave -> 16 out-channels x 32 pixels.
__global__ __launch_bounds__(128) void k_conv3x3(const float* __restrict__ x,
                                                 const bf16_t* __restrict__ w9,
                                                 float* __restrict__ y) {
  const int bh = blockIdx.x, b = bh / Hh, h = bh % Hh;
  const int w0 = blockIdx.y * 32;
  __shared__ __align__(16) bf16_t lds[3 * 34 * 64];  // [row][ww][c]
  for (int idx = threadIdx.x; idx < 3 * 34 * 64; idx += 128) {
    int c = idx / 102, rem = idx - c * 102, row = rem / 34, ww = rem % 34;
    int hh = h - 1 + row, wg = w0 - 1 + ww;
    float v = 0.f;
    if ((unsigned)hh < Hh && (unsigned)wg < Ww)
      v = x[((size_t)(b * Dd + c) * Hh + hh) * Ww + wg];
    lds[(row * 34 + ww) * 64 + c] = (bf16_t)v;
  }
  __syncthreads();
  const int wv = threadIdx.x >> 5, m0 = wv * 16;
  const int lane = threadIdx.x & 31, n = lane & 15, half = lane >> 4;
  v8f acc0 = {}, acc1 = {};
  for (int ky = 0; ky < 3; ++ky)
    for (int kx = 0; kx < 3; ++kx) {
      const bf16_t* wt = w9 + (ky * 3 + kx) * 4096 + m0 * 64;
      for (int kc = 0; kc < 64; kc += 32) {
        v16bf a  = load_frag_a(wt, 64, kc);
        v16bf b0 = load_frag_b(lds + (ky * 34 + kx) * 64, 64, kc);
        v16bf b1 = load_frag_b(lds + (ky * 34 + 16 + kx) * 64, 64, kc);
        acc0 = wmma_bf16(a, b0, acc0);
        acc1 = wmma_bf16(a, b1, acc1);
      }
    }
  for (int r = 0; r < 8; ++r) {
    int d = m0 + r + 8 * half;
    size_t base = ((size_t)(b * Dd + d) * Hh + h) * Ww + w0;
    y[base + n] = acc0[r];
    y[base + 16 + n] = acc1[r];
  }
}

// ---------------- batchnorm: two-stage stats + apply ----------------
// stage 1: grid (Dd, 16); each block reduces an 8192-element slice.
__global__ __launch_bounds__(256) void k_bnstats1(const float* __restrict__ x,
                                                  float* __restrict__ part) {
  const int d = blockIdx.x, sl = blockIdx.y;
  const int per = (Bb * HWp) / 16;  // 8192
  float s = 0.f, s2 = 0.f;
  for (int i = threadIdx.x; i < per; i += 256) {
    int nidx = sl * per + i;
    int b = nidx / HWp, pix = nidx % HWp;
    float v = x[(size_t)(b * Dd + d) * HWp + pix];
    s += v; s2 += v * v;
  }
  __shared__ float ls[256], ls2[256];
  ls[threadIdx.x] = s; ls2[threadIdx.x] = s2;
  __syncthreads();
  for (int o = 128; o; o >>= 1) {
    if (threadIdx.x < o) { ls[threadIdx.x] += ls[threadIdx.x + o];
                           ls2[threadIdx.x] += ls2[threadIdx.x + o]; }
    __syncthreads();
  }
  if (threadIdx.x == 0) {
    part[d * 16 + sl] = ls[0];
    part[1024 + d * 16 + sl] = ls2[0];
  }
}
// stage 2: 1 block, 64 threads; deterministic final combine.
__global__ __launch_bounds__(64) void k_bnstats2(const float* __restrict__ part,
                                                 float* __restrict__ stats) {
  const int d = threadIdx.x;
  float s = 0.f, s2 = 0.f;
  for (int i = 0; i < 16; ++i) { s += part[d * 16 + i]; s2 += part[1024 + d * 16 + i]; }
  const float invN = 1.f / (float)(Bb * HWp);
  float m = s * invN;
  float var = s2 * invN - m * m;
  stats[d] = m;
  stats[Dd + d] = rsqrtf(var + BN_EPS);
}

__global__ __launch_bounds__(256) void k_bnapply(const float* __restrict__ x,
                                                 const float* __restrict__ stats,
                                                 const float* __restrict__ g,
                                                 const float* __restrict__ be,
                                                 float* __restrict__ y) {
  size_t i = (size_t)blockIdx.x * 256 + threadIdx.x;
  int d = (int)((i / HWp) % Dd);
  y[i] = (x[i] - stats[d]) * stats[Dd + d] * g[d] + be[d];
}

// ---------------- 1x1 conv (per-pixel GEMM) -> bf16 output ----------------
// grid B*HW/64, 128 threads = 4 waves; tile = 64 pixels.
__global__ __launch_bounds__(128) void k_gemm1x1(const float* __restrict__ x,
                                                 const bf16_t* __restrict__ wm,
                                                 const float* __restrict__ bias,
                                                 bf16_t* __restrict__ y,
                                                 int Co, int Ci) {
  const int p0 = blockIdx.x * 64;
  const int b = p0 / HWp, pix0 = p0 % HWp;
  __shared__ __align__(16) bf16_t lds[64 * 64];  // [px][c]
  for (int idx = threadIdx.x; idx < Ci * 64; idx += 128) {
    int c = idx >> 6, px = idx & 63;
    lds[px * Ci + c] = (bf16_t)x[(size_t)(b * Ci + c) * HWp + pix0 + px];
  }
  __syncthreads();
  const int wv = threadIdx.x >> 5;
  const int lane = threadIdx.x & 31, n = lane & 15, half = lane >> 4;
  const int ntile = (Co >> 4) * 4;
  for (int t = wv; t < ntile; t += 4) {
    int mt = t >> 2, nt = t & 3;
    v8f acc = {};
    for (int kc = 0; kc < Ci; kc += 32) {
      v16bf a  = load_frag_a(wm + mt * 16 * Ci, Ci, kc);
      v16bf bb = load_frag_b(lds + (nt * 16) * Ci, Ci, kc);
      acc = wmma_bf16(a, bb, acc);
    }
    for (int r = 0; r < 8; ++r) {
      int co = mt * 16 + r + 8 * half;
      y[(size_t)(b * Co + co) * HWp + pix0 + nt * 16 + n] =
          (bf16_t)(acc[r] + bias[co]);
    }
  }
}

// ---------------- eH: per (b,w) column, Q^T(HxC) * K(CxH), diag mask -------
// logits written directly as bf16 (bf16 keeps f32 exponent range, mask survives)
__global__ __launch_bounds__(256) void k_eH(const bf16_t* __restrict__ q,
                                            const bf16_t* __restrict__ k,
                                            bf16_t* __restrict__ eH) {
  const int b = blockIdx.x / Ww, w = blockIdx.x % Ww;
  __shared__ __align__(16) bf16_t qs[Hh * 32], ks[Hh * 32];  // [h][c]
  for (int idx = threadIdx.x; idx < 32 * Hh; idx += 256) {
    int c = idx / Hh, hh = idx % Hh;
    size_t g = ((size_t)(b * Cc + c) * Hh + hh) * Ww + w;
    qs[hh * 32 + c] = q[g];
    ks[hh * 32 + c] = k[g];
  }
  __syncthreads();
  const int wv = threadIdx.x >> 5, h0 = wv * 16;
  const int lane = threadIdx.x & 31, n = lane & 15, half = lane >> 4;
  v16bf a = load_frag_a(qs + h0 * 32, 32, 0);
  for (int nt = 0; nt < 8; ++nt) {
    v16bf bb = load_frag_b(ks + nt * 16 * 32, 32, 0);
    v8f acc = {};
    acc = wmma_bf16(a, bb, acc);
    for (int r = 0; r < 8; ++r) {
      int h = h0 + r + 8 * half, g = nt * 16 + n;
      float v = (h == g) ? -3.0e38f : acc[r];
      eH[(((size_t)b * Ww + w) * Hh + h) * Hh + g] = (bf16_t)v;
    }
  }
}

// ---------------- eW: per (b,h) row, Q^T(WxC) * K(CxW) ----------------
__global__ __launch_bounds__(256) void k_eW(const bf16_t* __restrict__ q,
                                            const bf16_t* __restrict__ k,
                                            bf16_t* __restrict__ eW) {
  const int b = blockIdx.x / Hh, h = blockIdx.x % Hh;
  __shared__ __align__(16) bf16_t qs[Ww * 32], ks[Ww * 32];  // [w][c]
  for (int idx = threadIdx.x; idx < 32 * Ww; idx += 256) {
    int c = idx / Ww, ww = idx % Ww;
    size_t g = ((size_t)(b * Cc + c) * Hh + h) * Ww + ww;
    qs[ww * 32 + c] = q[g];
    ks[ww * 32 + c] = k[g];
  }
  __syncthreads();
  const int wv = threadIdx.x >> 5;
  const int lane = threadIdx.x & 31, n = lane & 15, half = lane >> 4;
  for (int mt = wv; mt < 16; mt += 8) {
    v16bf a = load_frag_a(qs + mt * 16 * 32, 32, 0);
    for (int nt = 0; nt < 16; ++nt) {
      v16bf bb = load_frag_b(ks + nt * 16 * 32, 32, 0);
      v8f acc = {};
      acc = wmma_bf16(a, bb, acc);
      for (int r = 0; r < 8; ++r) {
        int w = mt * 16 + r + 8 * half, g = nt * 16 + n;
        eW[(((size_t)b * Hh + h) * Ww + w) * Ww + g] = (bf16_t)acc[r];
      }
    }
  }
}

// ---------------- softmax over concat(H,W)=384, bf16 in / bf16 out ---------
// one wave per pixel, 8 waves/block; math in f32 registers.
__global__ __launch_bounds__(256) void k_softmax(bf16_t* __restrict__ eH,
                                                 bf16_t* __restrict__ eW) {
  const int p = blockIdx.x * 8 + (threadIdx.x >> 5);
  const int lane = threadIdx.x & 31;
  const int b = p / (Hh * Ww), rem = p % (Hh * Ww), h = rem / Ww, w = rem % Ww;
  bf16_t* rH = eH + (((size_t)b * Ww + w) * Hh + h) * Hh;
  bf16_t* rW = eW + (((size_t)b * Hh + h) * Ww + w) * Ww;
  float vH[4], vW[8];
  float mx = -3.4e38f;
  for (int i = 0; i < 4; ++i) { vH[i] = (float)rH[lane + 32 * i]; mx = fmaxf(mx, vH[i]); }
  for (int i = 0; i < 8; ++i) { vW[i] = (float)rW[lane + 32 * i]; mx = fmaxf(mx, vW[i]); }
  for (int o = 16; o; o >>= 1) mx = fmaxf(mx, __shfl_xor(mx, o, 32));
  float s = 0.f;
  for (int i = 0; i < 4; ++i) { vH[i] = __expf(vH[i] - mx); s += vH[i]; }
  for (int i = 0; i < 8; ++i) { vW[i] = __expf(vW[i] - mx); s += vW[i]; }
  for (int o = 16; o; o >>= 1) s += __shfl_xor(s, o, 32);
  const float inv = 1.f / s;
  for (int i = 0; i < 4; ++i) rH[lane + 32 * i] = (bf16_t)(vH[i] * inv);
  for (int i = 0; i < 8; ++i) rW[lane + 32 * i] = (bf16_t)(vW[i] * inv);
}

// ---------------- oH: per (b,w) column: V(DxH) * aH^T(HxH) ----------------
__global__ __launch_bounds__(256) void k_oH(const bf16_t* __restrict__ v,
                                            const bf16_t* __restrict__ aH,
                                            float* __restrict__ oH) {
  const int b = blockIdx.x / Ww, w = blockIdx.x % Ww;
  __shared__ __align__(16) bf16_t vs[Dd * Hh];  // [d][g]
  for (int idx = threadIdx.x; idx < Dd * Hh; idx += 256) {
    int d = idx >> 7, g = idx & 127;
    vs[d * Hh + g] = v[((size_t)(b * Dd + d) * Hh + g) * Ww + w];
  }
  __syncthreads();
  const int wv = threadIdx.x >> 5;
  const int lane = threadIdx.x & 31, n = lane & 15, half = lane >> 4;
  for (int t = wv; t < 32; t += 8) {
    int dt = t & 3, ht = t >> 2;
    v8f acc = {};
    const bf16_t* bbase = aH + (((size_t)b * Ww + w) * Hh + ht * 16) * Hh;
    for (int kc = 0; kc < Hh; kc += 32) {
      v16bf a  = load_frag_a(vs + dt * 16 * Hh, Hh, kc);
      v16bf bb = load_frag_b(bbase, Hh, kc);  // dense bf16 rows, stride 128
      acc = wmma_bf16(a, bb, acc);
    }
    for (int r = 0; r < 8; ++r) {
      int d = dt * 16 + r + 8 * half, h = ht * 16 + n;
      oH[((size_t)(b * Dd + d) * Hh + h) * Ww + w] = acc[r];
    }
  }
}

// ---------------- oW + combine: y = gamma*(oH+oW) + x ----------------
__global__ __launch_bounds__(256) void k_oW_combine(const bf16_t* __restrict__ v,
                                                    const bf16_t* __restrict__ aW,
                                                    const float* __restrict__ oH,
                                                    const float* __restrict__ xin,
                                                    const float* __restrict__ gptr,
                                                    float* __restrict__ y) {
  const int b = blockIdx.x / Hh, h = blockIdx.x % Hh;
  const float gm = *gptr;
  const int wv = threadIdx.x >> 5;
  const int lane = threadIdx.x & 31, n = lane & 15, half = lane >> 4;
  for (int t = wv; t < 64; t += 8) {
    int dt = t & 3, wt = t >> 2;
    v8f acc = {};
    const bf16_t* abase = v + ((size_t)(b * Dd + dt * 16) * Hh + h) * Ww;  // row-major
    const bf16_t* bbase = aW + (((size_t)b * Hh + h) * Ww + wt * 16) * Ww;
    for (int kc = 0; kc < Ww; kc += 32) {
      v16bf a  = load_frag_a(abase, HWp, kc);  // stride H*W between d-rows
      v16bf bb = load_frag_b(bbase, Ww, kc);   // dense bf16 rows, stride 256
      acc = wmma_bf16(a, bb, acc);
    }
    for (int r = 0; r < 8; ++r) {
      int d = dt * 16 + r + 8 * half, w = wt * 16 + n;
      size_t i = ((size_t)(b * Dd + d) * Hh + h) * Ww + w;
      y[i] = gm * (oH[i] + acc[r]) + xin[i];
    }
  }
}

// ---------------- driver ----------------
extern "C" void kernel_launch(void* const* d_in, const int* in_sizes, int n_in,
                              void* d_out, int out_size, void* d_ws, size_t ws_size,
                              hipStream_t stream) {
  (void)in_sizes; (void)n_in; (void)out_size; (void)ws_size;
  const float* cost = (const float*)d_in[0];
  const float* lf0  = (const float*)d_in[1];
  const float* lf1  = (const float*)d_in[2];
  const float* Wa   = (const float*)d_in[3];
  const float* ag   = (const float*)d_in[4];
  const float* ab   = (const float*)d_in[5];
  const float* Wq   = (const float*)d_in[6];
  const float* bq   = (const float*)d_in[7];
  const float* Wk   = (const float*)d_in[8];
  const float* bk   = (const float*)d_in[9];
  const float* Wv   = (const float*)d_in[10];
  const float* bv   = (const float*)d_in[11];
  const float* gmm  = (const float*)d_in[12];
  const float* Wb   = (const float*)d_in[13];
  const float* bg   = (const float*)d_in[14];
  const float* bbta = (const float*)d_in[15];

  char* ws = (char*)d_ws;
  float*  t0  = (float*)(ws + OFF_T0);
  float*  t1  = (float*)(ws + OFF_T1);
  float*  oHt = (float*)(ws + OFF_OH);
  bf16_t* eH  = (bf16_t*)(ws + OFF_EH);
  bf16_t* eW  = (bf16_t*)(ws + OFF_EW);
  bf16_t* qbf = (bf16_t*)(ws + OFF_Q);
  bf16_t* kbf = (bf16_t*)(ws + OFF_K);
  bf16_t* vbf = (bf16_t*)(ws + OFF_V);
  bf16_t* wa  = (bf16_t*)(ws + OFF_WA);
  bf16_t* wb  = (bf16_t*)(ws + OFF_WB);
  bf16_t* wq  = (bf16_t*)(ws + OFF_WQ);
  bf16_t* wk  = (bf16_t*)(ws + OFF_WK);
  bf16_t* wv  = (bf16_t*)(ws + OFF_WV);
  float*  st  = (float*)(ws + OFF_ST);
  float*  pt  = (float*)(ws + OFF_PT);

  // 0) weights -> bf16
  k_wconv<<<(79872 + 255) / 256, 256, 0, stream>>>(Wa, Wb, Wq, Wk, Wv,
                                                   wa, wb, wq, wk, wv);
  // 1) conv-a + BN
  k_conv3x3<<<dim3(Bb * Hh, Ww / 32), 128, 0, stream>>>(cost, wa, t0);
  k_bnstats1<<<dim3(Dd, 16), 256, 0, stream>>>(t0, pt);
  k_bnstats2<<<1, 64, 0, stream>>>(pt, st);
  k_bnapply<<<(Bb * Dd * HWp) / 256, 256, 0, stream>>>(t0, st, ag, ab, t0);
  // 2) q/k (shared across both recurrences) and attention map (computed once)
  k_gemm1x1<<<Bb * HWp / 64, 128, 0, stream>>>(lf0, wq, bq, qbf, Cc, Cc);
  k_gemm1x1<<<Bb * HWp / 64, 128, 0, stream>>>(lf1, wk, bk, kbf, Cc, Cc);
  k_eH<<<Bb * Ww, 256, 0, stream>>>(qbf, kbf, eH);
  k_eW<<<Bb * Hh, 256, 0, stream>>>(qbf, kbf, eW);
  k_softmax<<<Bb * HWp / 8, 256, 0, stream>>>(eH, eW);  // in-place bf16 probs
  // 3) recurrence 1: t0 -> t1
  k_gemm1x1<<<Bb * HWp / 64, 128, 0, stream>>>(t0, wv, bv, vbf, Dd, Dd);
  k_oH<<<Bb * Ww, 256, 0, stream>>>(vbf, eH, oHt);
  k_oW_combine<<<Bb * Hh, 256, 0, stream>>>(vbf, eW, oHt, t0, gmm, t1);
  // 4) recurrence 2: t1 -> t0
  k_gemm1x1<<<Bb * HWp / 64, 128, 0, stream>>>(t1, wv, bv, vbf, Dd, Dd);
  k_oH<<<Bb * Ww, 256, 0, stream>>>(vbf, eH, oHt);
  k_oW_combine<<<Bb * Hh, 256, 0, stream>>>(vbf, eW, oHt, t1, gmm, t0);
  // 5) conv-b + BN -> d_out
  k_conv3x3<<<dim3(Bb * Hh, Ww / 32), 128, 0, stream>>>(t0, wb, t1);
  k_bnstats1<<<dim3(Dd, 16), 256, 0, stream>>>(t1, pt);
  k_bnstats2<<<1, 64, 0, stream>>>(pt, st);
  k_bnapply<<<(Bb * Dd * HWp) / 256, 256, 0, stream>>>(t1, st, bg, bbta,
                                                       (float*)d_out);
}
